// MOEGate_78786880078245
// MI455X (gfx1250) — matrix operand order repaired and checked by reading, not compiled, not run on hardware
//
#include <hip/hip_runtime.h>

typedef __attribute__((ext_vector_type(16))) _Float16 v16h;
typedef __attribute__((ext_vector_type(8)))  float    v8f;

#define INPUT_DIM      512
#define HIDDEN         128
#define NEXP           10
#define ROWS_PER_BLOCK 128
#define WAVES          8
#define HSTRIDE        136   // f16 elements per H row in LDS (128 + 8 pad)

#define W1_FRAGS (16*8)      // 16 k-steps x 8 n-tiles
#define W2_FRAGS 4           // 4 k-steps x 1 n-tile

// ---------------------------------------------------------------------------
// Pack W1 [512,128] f32 and W2 [128,10] f32 into f16 WMMA B-fragment order.
// 16-bit B layout (ISA 7.12.2): per 32x16 fragment, lane l holds N = l&15,
// K = (l>>4)*8 + i for i < 8, K = 16 + (l>>4)*8 + i-8 for i >= 8 (16 f16 per lane).
// ---------------------------------------------------------------------------
__global__ void moe_pack_weights(const float* __restrict__ W1,
                                 const float* __restrict__ W2,
                                 _Float16* __restrict__ w1p,
                                 _Float16* __restrict__ w2p) {
  int t = blockIdx.x * blockDim.x + threadIdx.x;
  if (t < W1_FRAGS * 32) {
    int lane  = t & 31;
    int frag  = t >> 5;
    int nt    = frag & 7;
    int kstep = frag >> 3;
    int half  = lane >> 4;
    int n     = lane & 15;
    v16h v;
#pragma unroll
    for (int i = 0; i < 16; ++i) {
      int k = kstep * 32 + half * 8 + (i < 8 ? i : 8 + i);
      v[i] = (_Float16)W1[k * HIDDEN + nt * 16 + n];
    }
    *(v16h*)(w1p + (size_t)t * 16) = v;
  } else if (t < (W1_FRAGS + W2_FRAGS) * 32) {
    int u     = t - W1_FRAGS * 32;
    int lane  = u & 31;
    int kstep = u >> 5;
    int half  = lane >> 4;
    int n     = lane & 15;
    v16h v;
#pragma unroll
    for (int i = 0; i < 16; ++i) {
      int k = kstep * 32 + half * 8 + (i < 8 ? i : 8 + i);
      v[i] = (n < NEXP) ? (_Float16)W2[k * NEXP + n] : (_Float16)0.0f;
    }
    *(v16h*)(w2p + (size_t)u * 16) = v;
  }
}

// ---------------------------------------------------------------------------
// Main gating kernel: 8 wave32s per block, 16 rows per wave.
// ---------------------------------------------------------------------------
__global__ __launch_bounds__(256)
void moe_gate_kernel(const float* __restrict__ x,
                     const _Float16* __restrict__ w1p,
                     const _Float16* __restrict__ w2p,
                     const float* __restrict__ b1,
                     const float* __restrict__ b2,
                     float* __restrict__ out,
                     int B) {
  __shared__ _Float16 Hs[WAVES][16 * HSTRIDE];  // per-wave relu(x@W1+b1) tile (f16)
  __shared__ float    Ls[WAVES][16 * 16];       // per-wave logits tile

  const int tid  = threadIdx.x;
  const int wave = tid >> 5;
  const int lane = tid & 31;
  const int half = lane >> 4;   // 0: lanes 0-15, 1: lanes 16-31
  const int n    = lane & 15;   // N / M-within-half coordinate

  const int rowBase = blockIdx.x * ROWS_PER_BLOCK + wave * 16;
  const float* xrow = x + (size_t)(rowBase + n) * INPUT_DIM;  // A-matrix row = lane&15

  // ---------------- GEMM1: H[16,128] = relu(X[16,512] @ W1 + b1) ----------
  v8f acc[8] = {};  // 8 N-tiles of 16, f32 C/D layout (8 VGPRs each)

  for (int ks = 0; ks < 16; ++ks) {
    // A fragment (16-bit A layout, ISA 7.12.2):
    //   element i<8 : K = ks*32 + half*8 + i        (contiguous 8 floats)
    //   element i>=8: K = ks*32 + half*8 + 16 + i-8 (contiguous 8 floats)
    const float4* p0 = (const float4*)(xrow + ks * 32 + half * 8);
    const float4* p1 = (const float4*)(xrow + ks * 32 + half * 8 + 16);
    float4 q0 = p0[0], q1 = p0[1];
    float4 q2 = p1[0], q3 = p1[1];
    v16h a;
    a[0]  = (_Float16)q0.x; a[1]  = (_Float16)q0.y; a[2]  = (_Float16)q0.z; a[3]  = (_Float16)q0.w;
    a[4]  = (_Float16)q1.x; a[5]  = (_Float16)q1.y; a[6]  = (_Float16)q1.z; a[7]  = (_Float16)q1.w;
    a[8]  = (_Float16)q2.x; a[9]  = (_Float16)q2.y; a[10] = (_Float16)q2.z; a[11] = (_Float16)q2.w;
    a[12] = (_Float16)q3.x; a[13] = (_Float16)q3.y; a[14] = (_Float16)q3.z; a[15] = (_Float16)q3.w;

    if (ks + 1 < 16)  // gfx1250 global_prefetch_b8 for the next K chunk of x
      __builtin_prefetch(xrow + (ks + 1) * 32 + half * 8, 0, 3);

#pragma unroll
    for (int nt = 0; nt < 8; ++nt) {
      v16h bf = *(const v16h*)(w1p + (((size_t)ks * 8 + nt) * 32 + lane) * 16);
      acc[nt] = __builtin_amdgcn_wmma_f32_16x16x32_f16(
          false, a, false, bf, (short)0, acc[nt], false, false);
    }
  }

  // Bias + ReLU, spill H tile to LDS as f16 (C/D layout: lane holds col n,
  // rows half*8+v for v=0..7, per N-tile).
  _Float16* hrow = &Hs[wave][0];
#pragma unroll
  for (int nt = 0; nt < 8; ++nt) {
    int   col = nt * 16 + n;
    float bv  = b1[col];
#pragma unroll
    for (int v = 0; v < 8; ++v) {
      float hv = acc[nt][v] + bv;
      hrow[(half * 8 + v) * HSTRIDE + col] = (_Float16)(hv > 0.0f ? hv : 0.0f);
    }
  }

  // ---------------- GEMM2: logits[16,10] = H[16,128] @ W2 + b2 ------------
  float cinit = (n < NEXP) ? b2[n] : 0.0f;  // C depends only on N
  v8f acc2;
#pragma unroll
  for (int v = 0; v < 8; ++v) acc2[v] = cinit;

#pragma unroll
  for (int ks = 0; ks < 4; ++ks) {
    v16h a2;
    int kb = ks * 32 + half * 8;
#pragma unroll
    for (int i = 0; i < 8; ++i) a2[i]     = hrow[n * HSTRIDE + kb + i];
#pragma unroll
    for (int i = 0; i < 8; ++i) a2[8 + i] = hrow[n * HSTRIDE + kb + 16 + i];
    v16h b2f = *(const v16h*)(w2p + ((size_t)ks * 32 + lane) * 16);
    acc2 = __builtin_amdgcn_wmma_f32_16x16x32_f16(
        false, a2, false, b2f, (short)0, acc2, false, false);
  }

  // Stage logits through LDS so each row lands in one lane.
  float* lrow = &Ls[wave][0];
#pragma unroll
  for (int v = 0; v < 8; ++v)
    lrow[(half * 8 + v) * 16 + n] = acc2[v];

  // ---------------- softmax -> top2 -> softmax-of-2 epilogue --------------
  if (lane < 16) {
    int r = lane;
    float lg[NEXP];
#pragma unroll
    for (int j = 0; j < NEXP; ++j) lg[j] = lrow[r * 16 + j];

    float mx = lg[0];
#pragma unroll
    for (int j = 1; j < NEXP; ++j) mx = fmaxf(mx, lg[j]);
    float g[NEXP];
    float s = 0.0f;
#pragma unroll
    for (int j = 0; j < NEXP; ++j) { g[j] = __expf(lg[j] - mx); s += g[j]; }
    float inv = 1.0f / s;
#pragma unroll
    for (int j = 0; j < NEXP; ++j) g[j] *= inv;   // gate probabilities

    // top-1 (strict > keeps earliest index, matching jax.lax.top_k ties)
    int   i1 = 0;
    float g1 = g[0];
#pragma unroll
    for (int j = 1; j < NEXP; ++j) if (g[j] > g1) { g1 = g[j]; i1 = j; }
    // top-2 among the rest
    int   i2 = (i1 == 0) ? 1 : 0;
    float g2 = g[i2];
#pragma unroll
    for (int j = 0; j < NEXP; ++j)
      if (j != i1 && g[j] > g2) { g2 = g[j]; i2 = j; }

    // expert_weights = softmax([g1, g2])  (over the probabilities themselves)
    float e2  = __expf(g2 - g1);
    float w1v = 1.0f / (1.0f + e2);
    float w2v = e2 * w1v;

    size_t gr = (size_t)rowBase + r;
    out[2 * gr]     = w1v;
    out[2 * gr + 1] = w2v;
    size_t ib = (size_t)2 * B;
    out[ib + 2 * gr]     = (float)i1;
    out[ib + 2 * gr + 1] = (float)i2;
  }
}

// ---------------------------------------------------------------------------
extern "C" void kernel_launch(void* const* d_in, const int* in_sizes, int n_in,
                              void* d_out, int out_size, void* d_ws, size_t ws_size,
                              hipStream_t stream) {
  const float* x  = (const float*)d_in[0];
  const float* W1 = (const float*)d_in[1];
  const float* b1 = (const float*)d_in[2];
  const float* W2 = (const float*)d_in[3];
  const float* b2 = (const float*)d_in[4];
  // d_in[5] = top_k scalar (reference uses 2)

  int B = in_sizes[0] / INPUT_DIM;

  _Float16* w1p = (_Float16*)d_ws;                       // 16*8*32*16 f16 = 128 KB
  _Float16* w2p = w1p + (size_t)W1_FRAGS * 32 * 16;      // 4*32*16 f16 = 4 KB

  int packThreads = (W1_FRAGS + W2_FRAGS) * 32;
  moe_pack_weights<<<(packThreads + 255) / 256, 256, 0, stream>>>(W1, W2, w1p, w2p);

  int nblocks = B / ROWS_PER_BLOCK;
  moe_gate_kernel<<<nblocks, 256, 0, stream>>>(x, w1p, w2p, b1, b2, (float*)d_out, B);
}
